// SwinTransformerBlock_3813930959454
// MI455X (gfx1250) — compile-verified, matrix operand order checked by reading, not run on hardware
//
#include <hip/hip_runtime.h>
#include <hip/hip_bf16.h>
#include <stdint.h>

// ---------------------------------------------------------------- constants
#define SHIFT 3
constexpr int DDim = 14, HDim = 56, WDim = 56, CC = 192;
constexpr int NTOK = 343;          // 7*7*7 tokens per window
constexpr int NP   = 352;          // padded to 22*16
constexpr int NHd  = 6, HD = 32;   // heads, head dim
constexpr int NWIN = 256;          // B * 128 windows
constexpr int NWB  = 128;          // windows per batch
constexpr int MLPD = 768;
constexpr int NTOTTOK = 2 * 14 * 56 * 56;   // 87808 = 5488 * 16
constexpr float EPSV   = 1e-5f;
constexpr float SCALEV = 0.17677669529663687f;   // 32^-0.5

// ---------------------------------------------------------------- wmma glue
typedef float  v8f   __attribute__((ext_vector_type(8)));
typedef __bf16 v16bf __attribute__((ext_vector_type(16)));

#define WMMA_BF16(a, b, c) \
  __builtin_amdgcn_wmma_f32_16x16x32_bf16(false, (a), false, (b), (short)0, (c), false, false)

static __device__ __forceinline__ uint16_t f2bf(float f) {
  union { float f; uint32_t u; } x; x.f = f;
  uint32_t r = x.u + 0x7FFFu + ((x.u >> 16) & 1u);   // round-to-nearest-even
  return (uint16_t)(r >> 16);
}

union BF16x16 { uint4 q[2]; v16bf v; };

// loads bf16 elements p[0..7] and p[16..23] (the ISA's per-lane K striping)
static __device__ __forceinline__ v16bf ld16(const uint16_t* p) {
  BF16x16 u;
  u.q[0] = *reinterpret_cast<const uint4*>(p);
  u.q[1] = *reinterpret_cast<const uint4*>(p + 16);
  return u.v;
}

static __device__ __forceinline__ v8f vzero() {
  v8f z = {0.f, 0.f, 0.f, 0.f, 0.f, 0.f, 0.f, 0.f};
  return z;
}

// K=192 step: issue all 24 b128 loads as one clause, then 6 WMMAs (partial waits)
static __device__ __forceinline__ v8f gemm6(const uint16_t* arow, const uint16_t* brow,
                                            int koff, v8f acc) {
  v16bf A[6], B[6];
#pragma unroll
  for (int i = 0; i < 6; ++i) {
    A[i] = ld16(arow + i * 32 + koff);
    B[i] = ld16(brow + i * 32 + koff);
  }
#pragma unroll
  for (int i = 0; i < 6; ++i) acc = WMMA_BF16(A[i], B[i], acc);
  return acc;
}

// A register-resident variant (A reused across many N tiles)
static __device__ __forceinline__ v8f gemm6A(const v16bf* A, const uint16_t* brow,
                                             int koff, v8f acc) {
  v16bf B[6];
#pragma unroll
  for (int i = 0; i < 6; ++i) B[i] = ld16(brow + i * 32 + koff);
#pragma unroll
  for (int i = 0; i < 6; ++i) acc = WMMA_BF16(A[i], B[i], acc);
  return acc;
}

// ---------------------------------------------------------------- K0: W -> W^T bf16
__global__ void k_transpose_bf16(const float* __restrict__ W, uint16_t* __restrict__ WT,
                                 int K, int N) {
  int i = blockIdx.x * blockDim.x + threadIdx.x;
  if (i >= K * N) return;
  int n = i / K, k = i - n * K;
  WT[i] = f2bf(W[(size_t)k * N + n]);
}

// ---------------------------------------------------------------- K0b: expand rel-pos bias
// biasx[head][row][col] = rpb[RPI[row][col]][head]  (removes index math from attention)
__global__ void k_bias_expand(const float* __restrict__ rpb, float* __restrict__ biasx) {
  int i = blockIdx.x * blockDim.x + threadIdx.x;
  if (i >= NHd * NTOK * NTOK) return;
  int head = i / (NTOK * NTOK);
  int rem  = i - head * (NTOK * NTOK);
  int r = rem / NTOK, c = rem - r * NTOK;
  int d1 = r / 49, r1 = r - d1 * 49, h1 = r1 / 7, w1 = r1 - h1 * 7;
  int d2 = c / 49, r2 = c - d2 * 49, h2 = r2 / 7, w2 = r2 - h2 * 7;
  int idx = ((d1 - d2 + 6) * 13 + (h1 - h2 + 6)) * 13 + (w1 - w2 + 6);
  biasx[i] = rpb[idx * NHd + head];
}

// ---------------------------------------------------------------- K1: LN1 + shift + partition
__global__ __launch_bounds__(192) void k_ln1_partition(
    const float* __restrict__ x, const float* __restrict__ g, const float* __restrict__ bt,
    uint16_t* __restrict__ wins) {
  int wi = blockIdx.x / NTOK;
  int t  = blockIdx.x % NTOK;
  int c  = threadIdx.x;
  int b    = wi / NWB;
  int wrem = wi % NWB;
  int dw = wrem >> 6, hw = (wrem >> 3) & 7, ww = wrem & 7;
  int td = t / 49; int tr = t - td * 49; int th = tr / 7; int tw = tr - th * 7;
  int d = (dw * 7 + td + SHIFT) % DDim;
  int h = (hw * 7 + th + SHIFT) % HDim;
  int w = (ww * 7 + tw + SHIFT) % WDim;
  size_t base = ((size_t)((b * DDim + d) * HDim + h) * WDim + w) * CC;
  float v = x[base + c];

  __shared__ float s1[6], s2[6], mv[2];
  float sv = v, sq = v * v;
#pragma unroll
  for (int off = 16; off >= 1; off >>= 1) {
    sv += __shfl_xor(sv, off, 32);
    sq += __shfl_xor(sq, off, 32);
  }
  int lane = threadIdx.x & 31, wvid = threadIdx.x >> 5;
  if (lane == 0) { s1[wvid] = sv; s2[wvid] = sq; }
  __syncthreads();
  if (threadIdx.x == 0) {
    float a = 0.f, q = 0.f;
    for (int i = 0; i < 6; ++i) { a += s1[i]; q += s2[i]; }
    float mean = a / CC;
    float var  = q / CC - mean * mean;
    mv[0] = mean; mv[1] = rsqrtf(var + EPSV);
  }
  __syncthreads();
  float out = (v - mv[0]) * mv[1] * g[c] + bt[c];
  wins[((size_t)wi * NP + t) * CC + c] = f2bf(out);
}

// ---------------------------------------------------------------- K2: QKV GEMM (A resident)
__global__ __launch_bounds__(256) void k_qkv(
    const uint16_t* __restrict__ wins, const uint16_t* __restrict__ wT,
    const float* __restrict__ qkv_b, uint16_t* __restrict__ qkv) {
  int win = blockIdx.x;
  int tid = threadIdx.x, wave = tid >> 5, lane = tid & 31;
  int m15 = lane & 15, half = lane >> 4, koff = half * 8;
  const uint16_t* abase = wins + (size_t)win * NP * CC;

  for (int mt = wave; mt < 22; mt += 8) {
    const uint16_t* arow = abase + (size_t)(mt * 16 + m15) * CC;
    v16bf A[6];
#pragma unroll
    for (int kb = 0; kb < 6; ++kb) A[kb] = ld16(arow + kb * 32 + koff);
    int rows[8]; bool rv[8];
#pragma unroll
    for (int r = 0; r < 8; ++r) { rows[r] = mt * 16 + r + half * 8; rv[r] = rows[r] < NTOK; }

    for (int nt = 0; nt < 36; ++nt) {
      int j = nt * 16 + m15;                      // output column 0..575
      const uint16_t* brow = wT + (size_t)j * CC;
      if (nt + 1 < 36)
        __builtin_prefetch(wT + (size_t)(j + 16) * CC + koff, 0, 0);  // global_prefetch_b8
      v8f acc = gemm6A(A, brow, koff, vzero());
      int which = j / CC; int jj = j - which * CC;
      int head = jj >> 5; int hd = jj & 31;
      float bias  = qkv_b[j];
      float scale = (which == 0) ? SCALEV : 1.f;
      uint16_t* dst = qkv + (((size_t)win * NHd + head) * 3 + which) * NP * HD + hd;
#pragma unroll
      for (int r = 0; r < 8; ++r) {
        float v = rv[r] ? (acc[r] + bias) * scale : 0.f;   // zero padded rows
        dst[(size_t)rows[r] * HD] = f2bf(v);
      }
    }
  }
}

// ---------------------------------------------------------------- K3: flash attention
__global__ __launch_bounds__(256) void k_attn(
    const uint16_t* __restrict__ qkv, const float* __restrict__ biasx,
    const float* __restrict__ mask, uint16_t* __restrict__ out) {
  int win  = blockIdx.x / NHd;
  int head = blockIdx.x % NHd;
  const uint16_t* qb = qkv + (((size_t)win * NHd + head) * 3 + 0) * NP * HD;
  const uint16_t* kg = qb + (size_t)NP * HD;
  const uint16_t* vg = qb + (size_t)2 * NP * HD;

  __shared__ uint16_t kS[NP * HD];        // K rows   [row][hd]
  __shared__ uint16_t vT[HD * NP];        // V^T      [hd][row]
  __shared__ uint16_t stage[8][16 * 32];  // per-wave P staging

  int tid = threadIdx.x;
  for (int i = tid; i < NP * HD; i += 256) {
    kS[i] = kg[i];
    int row = i >> 5, hd = i & 31;
    vT[hd * NP + row] = vg[i];
  }
  __syncthreads();

  int wave = tid >> 5, lane = tid & 31;
  int m15 = lane & 15, half = lane >> 4, koff = half * 8;
  const float* mbase = mask + (size_t)(win % NWB) * NTOK * NTOK;
  const float* bbase = biasx + (size_t)head * NTOK * NTOK;

  for (int rt = wave; rt < 22; rt += 8) {
    v16bf qa = ld16(qb + (size_t)(rt * 16 + m15) * HD + koff);
    v8f o0 = vzero(), o1 = vzero();
    float mrow[8], lrow[8];
    bool rv[8]; int rowi[8];
    const float* bp[8]; const float* mp[8];
#pragma unroll
    for (int r = 0; r < 8; ++r) {
      mrow[r] = -3.0e38f; lrow[r] = 0.f;
      int row = rt * 16 + r + half * 8;
      rowi[r] = row; rv[r] = row < NTOK;
      int t1 = rv[r] ? row : 0;
      bp[r] = bbase + (size_t)t1 * NTOK;
      mp[r] = mbase + (size_t)t1 * NTOK;
    }

    for (int ctp = 0; ctp < 11; ++ctp) {
      int colb = ctp * 32;
      v16bf b0 = ld16(&kS[(colb + m15) * HD + koff]);
      v8f s0 = WMMA_BF16(qa, b0, vzero());
      v16bf b1 = ld16(&kS[(colb + 16 + m15) * HD + koff]);
      v8f s1 = WMMA_BF16(qa, b1, vzero());

      // precomputed rel-pos bias + shifted-window mask (+ padding mask)
#pragma unroll
      for (int sub = 0; sub < 2; ++sub) {
        int col = colb + sub * 16 + m15;
        bool cvalid = col < NTOK;
        int t2 = cvalid ? col : 0;
#pragma unroll
        for (int r = 0; r < 8; ++r) {
          float sv = (sub == 0) ? s0[r] : s1[r];
          if (cvalid && rv[r]) sv += bp[r][t2] + mp[r][t2];
          else                 sv = -1.0e30f;
          if (sub == 0) s0[r] = sv; else s1[r] = sv;
        }
      }

      // streaming softmax update (row reductions across the 16-lane half)
#pragma unroll
      for (int r = 0; r < 8; ++r) {
        float vmax = fmaxf(s0[r], s1[r]);
#pragma unroll
        for (int off = 8; off >= 1; off >>= 1) vmax = fmaxf(vmax, __shfl_xor(vmax, off, 16));
        float mnew = fmaxf(mrow[r], vmax);
        float sc   = __expf(mrow[r] - mnew);
        mrow[r]    = mnew;
        float p0 = __expf(s0[r] - mnew);
        float p1 = __expf(s1[r] - mnew);
        float rs = p0 + p1;
#pragma unroll
        for (int off = 8; off >= 1; off >>= 1) rs += __shfl_xor(rs, off, 16);
        lrow[r] = lrow[r] * sc + rs;
        o0[r] *= sc; o1[r] *= sc;
        int mr = r + half * 8;
        stage[wave][mr * 32 + m15]      = f2bf(p0);
        stage[wave][mr * 32 + 16 + m15] = f2bf(p1);
      }
      asm volatile("s_wait_dscnt 0" ::: "memory");   // cross-lane LDS RAW inside wave

      v16bf pa  = ld16(&stage[wave][m15 * 32 + koff]);
      v16bf vb0 = ld16(&vT[(0 * 16 + m15) * NP + colb + koff]);
      v16bf vb1 = ld16(&vT[(1 * 16 + m15) * NP + colb + koff]);
      o0 = WMMA_BF16(pa, vb0, o0);
      o1 = WMMA_BF16(pa, vb1, o1);
    }

#pragma unroll
    for (int r = 0; r < 8; ++r) {
      if (rv[r]) {
        float inv = 1.f / lrow[r];
        size_t ob = ((size_t)win * NP + rowi[r]) * CC + head * HD;
        out[ob + m15]      = f2bf(o0[r] * inv);
        out[ob + 16 + m15] = f2bf(o1[r] * inv);
      }
    }
  }
}

// ---------------------------------------------------------------- K4: proj + reverse + residual
__global__ __launch_bounds__(256) void k_proj(
    const uint16_t* __restrict__ ao, const uint16_t* __restrict__ wT,
    const float* __restrict__ pb, const float* __restrict__ x,
    float* __restrict__ xnew) {
  int win = blockIdx.x;
  int tid = threadIdx.x, wave = tid >> 5, lane = tid & 31;
  int m15 = lane & 15, half = lane >> 4, koff = half * 8;
  int b = win / NWB, wrem = win % NWB;
  int dw = wrem >> 6, hw = (wrem >> 3) & 7, ww = wrem & 7;
  const uint16_t* abase = ao + (size_t)win * NP * CC;

  for (int mt = wave; mt < 22; mt += 8) {
    const uint16_t* arow = abase + (size_t)(mt * 16 + m15) * CC;
    v16bf A[6];
#pragma unroll
    for (int kb = 0; kb < 6; ++kb) A[kb] = ld16(arow + kb * 32 + koff);

    // hoist window-reverse coordinate math out of the N loop
    size_t gib[8]; bool rv[8];
#pragma unroll
    for (int r = 0; r < 8; ++r) {
      int row = mt * 16 + r + half * 8;
      rv[r] = row < NTOK;
      int t = rv[r] ? row : 0;
      int td = t / 49, rr = t - td * 49, th = rr / 7, tw = rr - th * 7;
      int d = (dw * 7 + td + SHIFT) % DDim;
      int h = (hw * 7 + th + SHIFT) % HDim;
      int w = (ww * 7 + tw + SHIFT) % WDim;
      gib[r] = ((size_t)((b * DDim + d) * HDim + h) * WDim + w) * CC;
    }

    for (int nt = 0; nt < 12; ++nt) {
      int col = nt * 16 + m15;
      v8f acc = gemm6A(A, wT + (size_t)col * CC, koff, vzero());
      float bias = pb[col];
#pragma unroll
      for (int r = 0; r < 8; ++r) {
        if (rv[r]) {
          size_t gi = gib[r] + col;
          xnew[gi] = x[gi] + acc[r] + bias;
        }
      }
    }
  }
}

// ---------------------------------------------------------------- K5: LN3 + MLP (fused) + residual
__global__ __launch_bounds__(256) void k_mlp(
    const float* __restrict__ xnew, const float* __restrict__ g3, const float* __restrict__ b3,
    const uint16_t* __restrict__ w1T, const float* __restrict__ b1,
    const uint16_t* __restrict__ w2T, const float* __restrict__ b2,
    float* __restrict__ outp) {
  __shared__ uint16_t lnt[16 * CC];      // LN'd tokens, bf16
  __shared__ uint16_t hbuf[16 * MLPD];   // gelu(fc1) intermediate, bf16
  __shared__ float ps[256], ps2[256];
  __shared__ float mean[16], rstd[16];

  int tid = threadIdx.x;
  size_t tb = (size_t)blockIdx.x * 16;

  {   // LN statistics: 16 threads per token
    int tok = tid >> 4, ts = tid & 15;
    const float* xr = xnew + (tb + tok) * CC;
    float a = 0.f, q = 0.f;
#pragma unroll
    for (int j = 0; j < 12; ++j) {
      float v = xr[ts + j * 16];
      a += v; q += v * v;
    }
    ps[tok * 16 + ts] = a; ps2[tok * 16 + ts] = q;
  }
  __syncthreads();
  if (tid < 16) {
    float a = 0.f, q = 0.f;
    for (int i = 0; i < 16; ++i) { a += ps[tid * 16 + i]; q += ps2[tid * 16 + i]; }
    float m = a / CC, v = q / CC - m * m;
    mean[tid] = m; rstd[tid] = rsqrtf(v + EPSV);
  }
  __syncthreads();
  for (int i = tid; i < 16 * CC; i += 256) {
    int tok = i / CC, c = i - tok * CC;
    float v = (xnew[(tb + tok) * CC + c] - mean[tok]) * rstd[tok] * g3[c] + b3[c];
    lnt[i] = f2bf(v);
  }
  __syncthreads();

  int wave = tid >> 5, lane = tid & 31;
  int m15 = lane & 15, half = lane >> 4, koff = half * 8;

  // fc1 + exact GELU : LN tile register-resident per wave
  {
    v16bf A[6];
#pragma unroll
    for (int kb = 0; kb < 6; ++kb) A[kb] = ld16(&lnt[m15 * CC + kb * 32 + koff]);
    for (int nt = wave; nt < 48; nt += 8) {
      int col = nt * 16 + m15;
      v8f acc = gemm6A(A, w1T + (size_t)col * CC, koff, vzero());
      float bias = b1[col];
#pragma unroll
      for (int r = 0; r < 8; ++r) {
        int row = r + half * 8;
        float v = acc[r] + bias;
        v = 0.5f * v * (1.f + erff(v * 0.70710678f));
        hbuf[row * MLPD + col] = f2bf(v);
      }
    }
  }
  __syncthreads();

  // fc2 + residual (K=768 in 4 chunks of 6 K-slices, loads grouped per chunk)
  for (int nt = wave; nt < 12; nt += 8) {
    int col = nt * 16 + m15;
    const uint16_t* brow = w2T + (size_t)col * MLPD;
    v8f acc = vzero();
#pragma unroll
    for (int g = 0; g < 4; ++g)
      acc = gemm6(&hbuf[m15 * MLPD + g * 192], brow + g * 192, koff, acc);
    float bias = b2[col];
#pragma unroll
    for (int r = 0; r < 8; ++r) {
      int row = r + half * 8;
      size_t gi = (tb + row) * CC + col;
      outp[gi] = xnew[gi] + acc[r] + bias;
    }
  }
}

// ---------------------------------------------------------------- launch
extern "C" void kernel_launch(void* const* d_in, const int* in_sizes, int n_in,
                              void* d_out, int out_size, void* d_ws, size_t ws_size,
                              hipStream_t stream) {
  (void)in_sizes; (void)n_in; (void)out_size; (void)ws_size;
  const float* x      = (const float*)d_in[0];
  const float* n1g    = (const float*)d_in[1];
  const float* n1b    = (const float*)d_in[2];
  const float* qkv_w  = (const float*)d_in[3];
  const float* qkv_b  = (const float*)d_in[4];
  const float* rpb    = (const float*)d_in[5];
  const float* proj_w = (const float*)d_in[6];
  const float* proj_b = (const float*)d_in[7];
  const float* n3g    = (const float*)d_in[8];
  const float* n3b    = (const float*)d_in[9];
  const float* fc1_w  = (const float*)d_in[10];
  const float* fc1_b  = (const float*)d_in[11];
  const float* fc2_w  = (const float*)d_in[12];
  const float* fc2_b  = (const float*)d_in[13];
  const float* amask  = (const float*)d_in[14];
  float* outp = (float*)d_out;

  auto al = [](size_t v) { return (v + 255) & ~(size_t)255; };
  char* ws = (char*)d_ws;
  size_t offA  = 0;                                            // wins / attn-out (bf16)
  size_t offQ  = al(offA + 2ull * NWIN * NP * CC);             // q,k,v (bf16)
  size_t offX  = al(offQ + 2ull * NWIN * NHd * 3 * NP * HD);   // x after attn (f32)
  size_t offWq = al(offX + 4ull * NTOTTOK * CC);
  size_t offWp = al(offWq + 2ull * 576 * 192);
  size_t offW1 = al(offWp + 2ull * 192 * 192);
  size_t offW2 = al(offW1 + 2ull * 768 * 192);
  size_t offBX = al(offW2 + 2ull * 192 * 768);                 // expanded bias (f32)

  uint16_t* wsA  = (uint16_t*)(ws + offA);
  uint16_t* wsQ  = (uint16_t*)(ws + offQ);
  float*    wsX  = (float*)(ws + offX);
  uint16_t* wqT  = (uint16_t*)(ws + offWq);
  uint16_t* wpT  = (uint16_t*)(ws + offWp);
  uint16_t* w1T  = (uint16_t*)(ws + offW1);
  uint16_t* w2T  = (uint16_t*)(ws + offW2);
  float*    bx   = (float*)(ws + offBX);

  k_transpose_bf16<<<(576 * 192 + 255) / 256, 256, 0, stream>>>(qkv_w, wqT, 192, 576);
  k_transpose_bf16<<<(192 * 192 + 255) / 256, 256, 0, stream>>>(proj_w, wpT, 192, 192);
  k_transpose_bf16<<<(192 * 768 + 255) / 256, 256, 0, stream>>>(fc1_w, w1T, 192, 768);
  k_transpose_bf16<<<(768 * 192 + 255) / 256, 256, 0, stream>>>(fc2_w, w2T, 768, 192);
  k_bias_expand<<<(NHd * NTOK * NTOK + 255) / 256, 256, 0, stream>>>(rpb, bx);

  k_ln1_partition<<<NWIN * NTOK, 192, 0, stream>>>(x, n1g, n1b, wsA);
  k_qkv<<<NWIN, 256, 0, stream>>>(wsA, wqT, qkv_b, wsQ);
  k_attn<<<NWIN * NHd, 256, 0, stream>>>(wsQ, bx, amask, wsA);   // overwrites wins
  k_proj<<<NWIN, 256, 0, stream>>>(wsA, wpT, proj_b, x, wsX);
  k_mlp<<<NTOTTOK / 16, 256, 0, stream>>>(wsX, n3g, n3b, w1T, fc1_b, w2T, fc2_b, outp);
}
